// FractalAttention3D_85770496901350
// MI455X (gfx1250) — matrix operand order repaired and checked by reading, not compiled, not run on hardware
//
#include <hip/hip_runtime.h>
#include <hip/hip_bf16.h>

// ---------------------------------------------------------------------------
// FractalAttention3D for MI455X (gfx1250, wave32, WMMA bf16 16x16x32)
// ---------------------------------------------------------------------------

typedef __attribute__((ext_vector_type(16))) __bf16 v16bf;
typedef __attribute__((ext_vector_type(8)))  __bf16 v8bf;
typedef __attribute__((ext_vector_type(8)))  float  v8f;

#define NR     13824
#define ND_    1728
#define E_     512
#define A_DIM  256
#define H_DIM  128

// round-to-nearest-even f32 -> bf16 (stored as ushort)
__device__ __forceinline__ unsigned short f2bf(float f) {
  union { float f; unsigned u; } v; v.f = f;
  unsigned u = v.u;
  return (unsigned short)((u + 0x7FFFu + ((u >> 16) & 1u)) >> 16);
}

__device__ __forceinline__ v16bf load16(const unsigned short* p) {
  return *(const v16bf*)(const void*)p;
}
__device__ __forceinline__ v8bf load8(const unsigned short* p) {
  return *(const v8bf*)(const void*)p;
}

__device__ __forceinline__ float tanh_fast(float x) {
  float e = __expf(2.0f * x);
  return (e - 1.0f) / (e + 1.0f);
}

__device__ __forceinline__ v8f wmma_bf16(v16bf a, v16bf b, v8f c) {
  // (neg_a, A, neg_b, B, c_mod, C, reuse_a, reuse_b)
  return __builtin_amdgcn_wmma_f32_16x16x32_bf16(false, a, false, b,
                                                 (short)0, c, false, false);
}

// ---------------------------------------------------------------------------
// Prep kernels
// ---------------------------------------------------------------------------

// out[i] = bf16(a[i] + b[i])
__global__ void k_add_bf16(const float* __restrict__ a, const float* __restrict__ b,
                           unsigned short* __restrict__ o, int n) {
  int i = blockIdx.x * blockDim.x + threadIdx.x;
  if (i < n) o[i] = f2bf(a[i] + b[i]);
}

// W [512, N] f32 -> Wt [N, 512] bf16
__global__ void k_pack_wt(const float* __restrict__ W, unsigned short* __restrict__ Wt, int N) {
  int i = blockIdx.x * blockDim.x + threadIdx.x;
  if (i >= N * E_) return;
  int n = i / E_, k = i - n * E_;
  Wt[i] = f2bf(W[(size_t)k * N + n]);
}

// pooled [1728, 64] f32 -> dft [64, 1728] bf16
__global__ void k_pack_dft(const float* __restrict__ pd, unsigned short* __restrict__ dft) {
  int i = blockIdx.x * blockDim.x + threadIdx.x;
  if (i >= 64 * ND_) return;
  int n = i / ND_, d = i - n * ND_;
  dft[i] = f2bf(pd[(size_t)d * 64 + n]);
}

// ---------------------------------------------------------------------------
// Projection GEMM: out[m, col_off + n] = bf16( base[m,:] @ Wt[n,:] + bias[n] )
// One 16x16 output tile per wave; K = 512 in 16 WMMA steps.
// ---------------------------------------------------------------------------
__global__ void __launch_bounds__(256) k_proj(
    const unsigned short* __restrict__ Abf,   // [M, 512] bf16
    const unsigned short* __restrict__ Wt,    // [N, 512] bf16 (pre-transposed)
    const float* __restrict__ bias,           // [N] f32
    unsigned short* __restrict__ outp,        // [M, 512] bf16, packed columns
    int mtiles, int col_off) {
  int lane = threadIdx.x & 31;
  int wid  = threadIdx.x >> 5;
  int tid  = blockIdx.x * 8 + wid;
  int mt = tid % mtiles;
  int nt = tid / mtiles;
  int m0 = mt * 16, n0 = nt * 16;
  int row = lane & 15;   // A: M row / B: N column for this lane
  int kh  = lane >> 4;   // lane-half selects K sub-range

  v8f acc = {};
  const unsigned short* arow = Abf + (size_t)(m0 + row) * E_;
  const unsigned short* brow = Wt  + (size_t)(n0 + row) * E_;
#pragma unroll
  for (int c = 0; c < E_ / 32; ++c) {
    union { v16bf v; v8bf h[2]; } a;
    a.h[0] = load8(arow + c * 32 + kh * 16);       // K = c*32 + kh*16 .. +7
    a.h[1] = load8(arow + c * 32 + kh * 16 + 8);   // K = c*32 + kh*16+8 .. +15
    v16bf b = load16(brow + c * 32 + kh * 16);     // K = c*32 + kh*16 .. +15
    acc = wmma_bf16(a.v, b, acc);
  }
  float bn = bias[n0 + row];
  unsigned short* obase = outp + col_off + n0 + row;
#pragma unroll
  for (int v = 0; v < 8; ++v)
    obase[(size_t)(m0 + v + 8 * kh) * E_] = f2bf(acc[v] + bn);
}

// ---------------------------------------------------------------------------
// Fused attention: per wave = 16 range rows, loop domains in chunks of 32.
// Unnormalized-exp softmax (logits sigma ~8 -> exp well inside fp32 range),
// normalize once at the end. P*contrast is restaged via LDS into WMMA
// A-layout for the final GEMM against domain_flat^T.
// ---------------------------------------------------------------------------
__global__ void __launch_bounds__(256) k_attn(
    const unsigned short* __restrict__ rproj,  // [NR, 512] bf16: repr|con|off
    const unsigned short* __restrict__ dproj,  // [ND, 512] bf16: repr|con|off
    const unsigned short* __restrict__ dft,    // [64, ND]  bf16
    float* __restrict__ out) {                 // [NR, 64]  f32
  __shared__ __align__(32) unsigned short smem[8 * 512];  // 1KB per wave
  int lane = threadIdx.x & 31;
  int wid  = threadIdx.x >> 5;
  int r0   = (blockIdx.x * 8 + wid) * 16;
  int row  = lane & 15;
  int kh   = lane >> 4;
  unsigned short* lds = smem + wid * 512;

  const float logit_scale = 0.25f;           // 1/(sqrt(256)*0.25)
  const float inv_h       = 0.0883883476f;   // 1/sqrt(128)

  // Resident A tiles for this wave's 16 range rows
  v16bf a_repr[8], a_con[4], a_off[4];
  const unsigned short* arow = rproj + (size_t)(r0 + row) * E_;
#pragma unroll
  for (int c = 0; c < 8; ++c) {
    union { v16bf v; v8bf h[2]; } t;
    t.h[0] = load8(arow + c * 32 + kh * 16);
    t.h[1] = load8(arow + c * 32 + kh * 16 + 8);
    a_repr[c] = t.v;
  }
#pragma unroll
  for (int c = 0; c < 4; ++c) {
    union { v16bf v; v8bf h[2]; } t;
    t.h[0] = load8(arow + A_DIM + c * 32 + kh * 16);
    t.h[1] = load8(arow + A_DIM + c * 32 + kh * 16 + 8);
    a_con[c] = t.v;
    union { v16bf v; v8bf h[2]; } u;
    u.h[0] = load8(arow + A_DIM + H_DIM + c * 32 + kh * 16);
    u.h[1] = load8(arow + A_DIM + H_DIM + c * 32 + kh * 16 + 8);
    a_off[c] = u.v;
  }

  v8f   acc[4] = {};     // 16x64 f32 output accumulator (unnormalized)
  float lsum[8]   = {};  // per-lane partial exp sums
  float offacc[8] = {};  // per-lane partial exp*offset sums

  for (int dbase = 0; dbase < ND_; dbase += 32) {
    if (dbase + 32 < ND_) {  // stream next chunk's domain rows into cache
      __builtin_prefetch(dproj + (size_t)(dbase + 32 + row) * E_, 0, 1);
      __builtin_prefetch(dproj + (size_t)(dbase + 48 + row) * E_, 0, 1);
    }
#pragma unroll
    for (int t = 0; t < 2; ++t) {
      const unsigned short* drow = dproj + (size_t)(dbase + t * 16 + row) * E_;
      // contrast (K=128)
      v8f Cm = {};
#pragma unroll
      for (int c = 0; c < 4; ++c)
        Cm = wmma_bf16(a_con[c], load16(drow + A_DIM + c * 32 + kh * 16), Cm);
      float ct[8];
#pragma unroll
      for (int v = 0; v < 8; ++v) ct[v] = tanh_fast(Cm[v] * inv_h) * 1.8f;
      // offset (K=128)
      v8f Om = {};
#pragma unroll
      for (int c = 0; c < 4; ++c)
        Om = wmma_bf16(a_off[c], load16(drow + A_DIM + H_DIM + c * 32 + kh * 16), Om);
      float of[8];
#pragma unroll
      for (int v = 0; v < 8; ++v) of[v] = tanh_fast(Om[v] * inv_h);
      // logits (K=256)
      v8f Sm = {};
#pragma unroll
      for (int c = 0; c < 8; ++c)
        Sm = wmma_bf16(a_repr[c], load16(drow + c * 32 + kh * 16), Sm);
#pragma unroll
      for (int v = 0; v < 8; ++v) {
        float p = __expf(Sm[v] * logit_scale);
        lsum[v]   += p;
        offacc[v] += p * of[v];
        // stage P*contrast: LDS tile row = v + 8*kh, col = t*16 + row
        lds[(v + 8 * kh) * 32 + t * 16 + row] = f2bf(p * ct[v]);
      }
    }
    asm volatile("s_wait_dscnt 0" ::: "memory");  // same-wave LDS store->load
    union { v16bf v; v8bf h[2]; } pa;             // re-read in A-layout (16x32)
    pa.h[0] = load8(lds + row * 32 + kh * 16);
    pa.h[1] = load8(lds + row * 32 + kh * 16 + 8);
#pragma unroll
    for (int nt = 0; nt < 4; ++nt) {
      const unsigned short* bdf = dft + (size_t)(nt * 16 + row) * ND_ + dbase + kh * 16;
      acc[nt] = wmma_bf16(pa.v, load16(bdf), acc[nt]);
    }
    asm volatile("s_wait_dscnt 0" ::: "memory");  // keep stores of next iter below loads
  }

  // reduce row sums across the 16 lanes of each half-wave
#pragma unroll
  for (int v = 0; v < 8; ++v) {
    float L = lsum[v], F = offacc[v];
    for (int m = 1; m < 16; m <<= 1) {
      L += __shfl_xor(L, m, 32);
      F += __shfl_xor(F, m, 32);
    }
    lsum[v] = L; offacc[v] = F;
  }

  // out = (accP + offacc) / lsum   (softmax normalization, fused)
#pragma unroll
  for (int nt = 0; nt < 4; ++nt)
#pragma unroll
    for (int v = 0; v < 8; ++v)
      out[(size_t)(r0 + v + 8 * kh) * 64 + nt * 16 + row] =
          (acc[nt][v] + offacc[v]) / lsum[v];
}

// ---------------------------------------------------------------------------
// Launch
// ---------------------------------------------------------------------------
extern "C" void kernel_launch(void* const* d_in, const int* in_sizes, int n_in,
                              void* d_out, int out_size, void* d_ws, size_t ws_size,
                              hipStream_t stream) {
  (void)in_sizes; (void)n_in; (void)out_size; (void)ws_size;
  const float* pooled  = (const float*)d_in[0];
  const float* r_lat   = (const float*)d_in[1];
  const float* d_lat   = (const float*)d_in[2];
  const float* r_pos   = (const float*)d_in[3];
  const float* d_pos   = (const float*)d_in[4];
  const float* W_rproj = (const float*)d_in[5];
  const float* b_rproj = (const float*)d_in[6];
  const float* W_dproj = (const float*)d_in[7];
  const float* b_dproj = (const float*)d_in[8];
  const float* W_rcon  = (const float*)d_in[9];
  const float* b_rcon  = (const float*)d_in[10];
  const float* W_dcon  = (const float*)d_in[11];
  const float* b_dcon  = (const float*)d_in[12];
  const float* W_roff  = (const float*)d_in[13];
  const float* b_roff  = (const float*)d_in[14];
  const float* W_doff  = (const float*)d_in[15];
  const float* b_doff  = (const float*)d_in[16];

  // workspace layout (bytes); total ~31.6 MiB
  char* ws = (char*)d_ws;
  unsigned short* base_r = (unsigned short*)(ws);              // NR*512 bf16
  unsigned short* base_d = (unsigned short*)(ws + 14155776);   // ND*512 bf16
  unsigned short* wt     = (unsigned short*)(ws + 15925248);   // packed weights
  unsigned short* rout   = (unsigned short*)(ws + 16973824);   // NR*512 bf16
  unsigned short* doutp  = (unsigned short*)(ws + 31129600);   // ND*512 bf16
  unsigned short* dft    = (unsigned short*)(ws + 32899072);   // 64*ND bf16

  unsigned short* wt_rproj = wt;
  unsigned short* wt_dproj = wt + 256 * E_;
  unsigned short* wt_rcon  = wt + 512 * E_;
  unsigned short* wt_dcon  = wt + 640 * E_;
  unsigned short* wt_roff  = wt + 768 * E_;
  unsigned short* wt_doff  = wt + 896 * E_;

  int n1 = NR  * E_;
  int n2 = ND_ * E_;
  k_add_bf16<<<(n1 + 255) / 256, 256, 0, stream>>>(r_lat, r_pos, base_r, n1);
  k_add_bf16<<<(n2 + 255) / 256, 256, 0, stream>>>(d_lat, d_pos, base_d, n2);

  k_pack_wt<<<(256 * E_ + 255) / 256, 256, 0, stream>>>(W_rproj, wt_rproj, 256);
  k_pack_wt<<<(256 * E_ + 255) / 256, 256, 0, stream>>>(W_dproj, wt_dproj, 256);
  k_pack_wt<<<(128 * E_ + 255) / 256, 256, 0, stream>>>(W_rcon,  wt_rcon,  128);
  k_pack_wt<<<(128 * E_ + 255) / 256, 256, 0, stream>>>(W_dcon,  wt_dcon,  128);
  k_pack_wt<<<(128 * E_ + 255) / 256, 256, 0, stream>>>(W_roff,  wt_roff,  128);
  k_pack_wt<<<(128 * E_ + 255) / 256, 256, 0, stream>>>(W_doff,  wt_doff,  128);
  k_pack_dft<<<(64 * ND_ + 255) / 256, 256, 0, stream>>>(pooled, dft);

  // projections: grid = mtiles*ntiles/8 waves-per-block
  k_proj<<<864 * 16 / 8, 256, 0, stream>>>(base_r, wt_rproj, b_rproj, rout, 864, 0);
  k_proj<<<864 *  8 / 8, 256, 0, stream>>>(base_r, wt_rcon,  b_rcon,  rout, 864, 256);
  k_proj<<<864 *  8 / 8, 256, 0, stream>>>(base_r, wt_roff,  b_roff,  rout, 864, 384);
  k_proj<<<108 * 16 / 8, 256, 0, stream>>>(base_d, wt_dproj, b_dproj, doutp, 108, 0);
  k_proj<<<108 *  8 / 8, 256, 0, stream>>>(base_d, wt_dcon,  b_dcon,  doutp, 108, 256);
  k_proj<<<108 *  8 / 8, 256, 0, stream>>>(base_d, wt_doff,  b_doff,  doutp, 108, 384);

  // fused attention: 864 waves (16 range rows each), 8 waves/block
  k_attn<<<108, 256, 0, stream>>>(rout, doutp, dft, (float*)d_out);
}
